// GAT_36636071035107
// MI455X (gfx1250) — compile-verified, matrix operand order checked by reading, not run on hardware
//
#include <hip/hip_runtime.h>
#include <hip/hip_bf16.h>

// ---------------------------------------------------------------------------
// GAT forward for MI455X (gfx1250, wave32, WMMA 16x16x32 f16->f32)
// N=4096, NFEAT=512, NHID=64, NCLASS=16, H1=8
// ---------------------------------------------------------------------------

#define N      4096
#define NFEAT  512
#define NHID   64
#define NCLASS 16
#define H1     8
#define NEG_SLOPE 0.2f
#define MASK_VAL  -1e10f

typedef __attribute__((ext_vector_type(16))) _Float16 v16h;
typedef __attribute__((ext_vector_type(8)))  _Float16 v8h;
typedef __attribute__((ext_vector_type(8)))  float    v8f;

// ---------------------------------------------------------------------------
// WMMA helpers
// ---------------------------------------------------------------------------
__device__ __forceinline__ v8f wmma16(v16h a, v16h b, v8f c) {
  // D = A(16x32 f16) x B(32x16 f16) + C(16x16 f32)
  return __builtin_amdgcn_wmma_f32_16x16x32_f16(
      /*neg_a=*/false, a, /*neg_b=*/false, b,
      /*c_mod=*/(short)0, c, /*reuse_a=*/false, /*reuse_b=*/false);
}

// Load a 16x32 (f16) A-fragment from a row-major matrix with leading dim `ld`.
// Also serves as the B-fragment loader when B is stored transposed ([N][K]),
// since the B lane layout mirrors A with N in place of M.
// Per ISA 7.12.2 (16-bit A 16x32): lanes 0-15 hold M=lane, K = {0..7,16..23};
// lanes 16-31 hold M=lane-16, K = {8..15,24..31}.  Both K runs are contiguous
// 8-half (16-byte) chunks -> two vector loads.
__device__ __forceinline__ v16h load_frag16(const _Float16* base, int rc,
                                            int ld, int k0) {
  int lane = threadIdx.x & 31;
  int hf   = lane >> 4;        // which half of the wave
  int r    = lane & 15;        // row (A) / column (B)
  const _Float16* p = base + (size_t)(rc + r) * ld + k0 + (hf << 3);
  v8h lo = *(const v8h*)p;          // K = k0 + 8*hf + [0..7]
  v8h hi = *(const v8h*)(p + 16);   // K = k0 + 16 + 8*hf + [0..7]
  return __builtin_shufflevector(lo, hi, 0, 1, 2, 3, 4, 5, 6, 7, 8, 9, 10, 11,
                                 12, 13, 14, 15);
}

__device__ __forceinline__ float lrelu(float x) {
  return x > 0.0f ? x : NEG_SLOPE * x;
}

// ---------------------------------------------------------------------------
// Kernel 0: precision casts + transposes
//   XH  [N][NFEAT]        f16  (row major, GEMM1 A)
//   W1T [H1][NHID][NFEAT] f16  (B^T: contiguous in K)
//   W2T [NCLASS][H1*NHID] f16
// ---------------------------------------------------------------------------
__global__ void cast_kernel(const float* __restrict__ x,
                            const float* __restrict__ W1,
                            const float* __restrict__ W2,
                            _Float16* __restrict__ XH,
                            _Float16* __restrict__ W1T,
                            _Float16* __restrict__ W2T) {
  int i = blockIdx.x * blockDim.x + threadIdx.x;
  if (i < N * NFEAT) XH[i] = (_Float16)x[i];
  if (i < H1 * NFEAT * NHID) {
    int h   = i / (NFEAT * NHID);
    int rem = i % (NFEAT * NHID);
    int k   = rem / NHID;  // input feature
    int c   = rem % NHID;  // output feature
    W1T[((size_t)(h * NHID + c)) * NFEAT + k] = (_Float16)W1[i];
  }
  if (i < (H1 * NHID) * NCLASS) {
    int k = i / NCLASS;
    int c = i % NCLASS;
    W2T[(size_t)c * (H1 * NHID) + k] = (_Float16)W2[i];
  }
}

// ---------------------------------------------------------------------------
// Kernel 1: feats1[h] = XH @ W1[h]   (per head, M=4096 N=64 K=512)
// One wave per 16-row tile; 4 accumulators cover all 64 output features.
// Writes f32 (for s/n dots) and transposed f16 [h][f][row] (attn-GEMM B).
// ---------------------------------------------------------------------------
__global__ void gemm1_kernel(const _Float16* __restrict__ XH,
                             const _Float16* __restrict__ W1T,
                             float* __restrict__ feats,
                             _Float16* __restrict__ featsT) {
  int head = blockIdx.y;
  int r0   = blockIdx.x << 4;
  const _Float16* BT = W1T + (size_t)head * NHID * NFEAT;

  v8f c0 = {}, c1 = {}, c2 = {}, c3 = {};
  for (int k0 = 0; k0 < NFEAT; k0 += 32) {
    v16h a = load_frag16(XH, r0, NFEAT, k0);
    c0 = wmma16(a, load_frag16(BT, 0, NFEAT, k0), c0);
    c1 = wmma16(a, load_frag16(BT, 16, NFEAT, k0), c1);
    c2 = wmma16(a, load_frag16(BT, 32, NFEAT, k0), c2);
    c3 = wmma16(a, load_frag16(BT, 48, NFEAT, k0), c3);
  }

  int lane = threadIdx.x & 31;
  int hf   = lane >> 4;
  int r    = lane & 15;
  v8f acc[4] = {c0, c1, c2, c3};
#pragma unroll
  for (int nt = 0; nt < 4; ++nt) {
#pragma unroll
    for (int k = 0; k < 8; ++k) {
      int orow = r0 + k + (hf << 3);
      int col  = nt * 16 + r;
      float v  = acc[nt][k];
      feats[((size_t)head * N + orow) * NHID + col] = v;
      featsT[((size_t)head * NHID + col) * N + orow] = (_Float16)v;
    }
  }
}

// ---------------------------------------------------------------------------
// Kernel 2: per-row attention dots  s[i] = feats[i]·a_s,  n[i] = feats[i]·a_n
// ---------------------------------------------------------------------------
__global__ void sn_kernel(const float* __restrict__ feats,
                          const float* __restrict__ a_s,
                          const float* __restrict__ a_n,
                          float* __restrict__ s, float* __restrict__ n,
                          int fout, int rowsPerHead, int total) {
  int i = blockIdx.x * blockDim.x + threadIdx.x;
  if (i >= total) return;
  int head = i / rowsPerHead;
  const float* f  = feats + (size_t)i * fout;
  const float* av = a_s + head * fout;
  const float* nv = a_n + head * fout;
  float ss = 0.0f, nn = 0.0f;
  for (int c = 0; c < fout; ++c) {
    float v = f[c];
    ss += v * av[c];
    nn += v * nv[c];
  }
  s[i] = ss;
  n[i] = nn;
}

// ---------------------------------------------------------------------------
// Kernel 3: softmax row statistics (max m, normalizer Z) per head per row.
// logit = adj>0 ? leaky_relu(s_i + n_j) : -1e10  (exactly like the reference)
// One block per row; adj row lives in L2 (adj = 64MB < 192MB L2).
// ---------------------------------------------------------------------------
__global__ void rowstats_kernel(const int* __restrict__ adj,
                                const float* __restrict__ s,
                                const float* __restrict__ n,
                                float* __restrict__ m, float* __restrict__ Z,
                                int nheads) {
  int i   = blockIdx.x;
  int tid = threadIdx.x;
  __shared__ float red[256];
  const int* arow = adj + (size_t)i * N;

  for (int h = 0; h < nheads; ++h) {
    float sv = s[h * N + i];
    const float* nrow = n + h * N;

    float lm = -3.0e38f;
    for (int j = tid; j < N; j += 256) {
      float lg = arow[j] > 0 ? lrelu(sv + nrow[j]) : MASK_VAL;
      lm = fmaxf(lm, lg);
    }
    red[tid] = lm;
    __syncthreads();
    for (int st = 128; st > 0; st >>= 1) {
      if (tid < st) red[tid] = fmaxf(red[tid], red[tid + st]);
      __syncthreads();
    }
    float mv = red[0];
    __syncthreads();

    float ls = 0.0f;
    for (int j = tid; j < N; j += 256) {
      float lg = arow[j] > 0 ? lrelu(sv + nrow[j]) : MASK_VAL;
      ls += __expf(lg - mv);
    }
    red[tid] = ls;
    __syncthreads();
    for (int st = 128; st > 0; st >>= 1) {
      if (tid < st) red[tid] += red[tid + st];
      __syncthreads();
    }
    if (tid == 0) {
      m[h * N + i] = mv;
      Z[h * N + i] = red[0];
    }
    __syncthreads();
  }
}

// ---------------------------------------------------------------------------
// Kernel 4: layer-1 fused attention GEMM:  out = softmax(logits) @ feats + b1
// Probability tiles are synthesized directly in WMMA A-fragment layout.
// relu applied; (elu o relu == relu) so this is the final layer-1 activation.
// Output written as f16 H[row][head*64+f] for the layer-2 GEMM.
// ---------------------------------------------------------------------------
__global__ void attn_gemm1_kernel(const int* __restrict__ adj,
                                  const _Float16* __restrict__ featsT,
                                  const float* __restrict__ s1,
                                  const float* __restrict__ n1,
                                  const float* __restrict__ m1,
                                  const float* __restrict__ Z1,
                                  const float* __restrict__ b1,
                                  _Float16* __restrict__ H) {
  int head = blockIdx.y;
  int r0   = blockIdx.x << 4;
  int lane = threadIdx.x & 31;
  int hf   = lane >> 4;
  int r    = lane & 15;
  int row  = r0 + r;

  const float* nrow = n1 + head * N;
  float sv = s1[head * N + row];
  float mv = m1[head * N + row];
  float zi = 1.0f / Z1[head * N + row];
  const int* arow = adj + (size_t)row * N;
  const _Float16* BT = featsT + (size_t)head * NHID * N;

  v8f c0 = {}, c1 = {}, c2 = {}, c3 = {};
  for (int j0 = 0; j0 < N; j0 += 32) {
    if (j0 + 256 < N) __builtin_prefetch(arow + j0 + 256, 0, 0);
    v16h a;
#pragma unroll
    for (int e = 0; e < 16; ++e) {
      int j = j0 + (e & 7) + (hf << 3) + ((e >> 3) << 4);
      float lg = lrelu(sv + nrow[j]);
      float p  = arow[j] > 0 ? __expf(lg - mv) * zi : 0.0f;
      a[e] = (_Float16)p;
    }
    c0 = wmma16(a, load_frag16(BT, 0, N, j0), c0);
    c1 = wmma16(a, load_frag16(BT, 16, N, j0), c1);
    c2 = wmma16(a, load_frag16(BT, 32, N, j0), c2);
    c3 = wmma16(a, load_frag16(BT, 48, N, j0), c3);
  }

  v8f acc[4] = {c0, c1, c2, c3};
#pragma unroll
  for (int nt = 0; nt < 4; ++nt) {
#pragma unroll
    for (int k = 0; k < 8; ++k) {
      int orow = r0 + k + (hf << 3);
      int col  = nt * 16 + r;
      float v  = acc[nt][k] + b1[head * NHID + col];
      v = v > 0.0f ? v : 0.0f;  // relu; elu(relu(x)) == relu(x)
      H[(size_t)orow * (H1 * NHID) + head * NHID + col] = (_Float16)v;
    }
  }
}

// ---------------------------------------------------------------------------
// Kernel 5: feats2 = H @ W2   (M=4096, N=16, K=512)
// ---------------------------------------------------------------------------
__global__ void gemm2_kernel(const _Float16* __restrict__ H,
                             const _Float16* __restrict__ W2T,
                             float* __restrict__ feats2,
                             _Float16* __restrict__ feats2T) {
  int r0 = blockIdx.x << 4;
  v8f c = {};
  for (int k0 = 0; k0 < H1 * NHID; k0 += 32) {
    v16h a = load_frag16(H, r0, H1 * NHID, k0);
    v16h b = load_frag16(W2T, 0, H1 * NHID, k0);
    c = wmma16(a, b, c);
  }
  int lane = threadIdx.x & 31;
  int hf   = lane >> 4;
  int r    = lane & 15;
#pragma unroll
  for (int k = 0; k < 8; ++k) {
    int orow = r0 + k + (hf << 3);
    float v  = c[k];
    feats2[(size_t)orow * NCLASS + r] = v;
    feats2T[(size_t)r * N + orow]     = (_Float16)v;
  }
}

// ---------------------------------------------------------------------------
// Kernel 6: layer-2 fused attention GEMM + bias + relu + row log-softmax.
// Classes sit across lanes (0-15 / 16-31), so the 16-wide log-softmax
// reductions are __shfl_xor steps inside the wave32.
// ---------------------------------------------------------------------------
__global__ void attn_gemm2_kernel(const int* __restrict__ adj,
                                  const _Float16* __restrict__ feats2T,
                                  const float* __restrict__ s2,
                                  const float* __restrict__ n2,
                                  const float* __restrict__ m2,
                                  const float* __restrict__ Z2,
                                  const float* __restrict__ b2,
                                  float* __restrict__ out) {
  int r0   = blockIdx.x << 4;
  int lane = threadIdx.x & 31;
  int hf   = lane >> 4;
  int r    = lane & 15;
  int row  = r0 + r;

  float sv = s2[row];
  float mv = m2[row];
  float zi = 1.0f / Z2[row];
  const int* arow = adj + (size_t)row * N;

  v8f c = {};
  for (int j0 = 0; j0 < N; j0 += 32) {
    if (j0 + 256 < N) __builtin_prefetch(arow + j0 + 256, 0, 0);
    v16h a;
#pragma unroll
    for (int e = 0; e < 16; ++e) {
      int j = j0 + (e & 7) + (hf << 3) + ((e >> 3) << 4);
      float lg = lrelu(sv + n2[j]);
      float p  = arow[j] > 0 ? __expf(lg - mv) * zi : 0.0f;
      a[e] = (_Float16)p;
    }
    v16h b = load_frag16(feats2T, 0, N, j0);
    c = wmma16(a, b, c);
  }

  float bias = b2[r];
#pragma unroll
  for (int k = 0; k < 8; ++k) {
    float v = c[k] + bias;
    v = v > 0.0f ? v : 0.0f;  // relu
    // log-softmax over the 16 classes (lanes within each 16-lane half)
    float mx = v;
#pragma unroll
    for (int msk = 1; msk < 16; msk <<= 1)
      mx = fmaxf(mx, __shfl_xor(mx, msk, 32));
    float ex = __expf(v - mx);
    float sum = ex;
#pragma unroll
    for (int msk = 1; msk < 16; msk <<= 1) sum += __shfl_xor(sum, msk, 32);
    float o  = (v - mx) - logf(sum);
    int orow = r0 + k + (hf << 3);
    out[(size_t)orow * NCLASS + r] = o;
  }
}

// ---------------------------------------------------------------------------
// Host launcher
// ---------------------------------------------------------------------------
extern "C" void kernel_launch(void* const* d_in, const int* in_sizes, int n_in,
                              void* d_out, int out_size, void* d_ws,
                              size_t ws_size, hipStream_t stream) {
  const float* x   = (const float*)d_in[0];
  const int*   adj = (const int*)d_in[1];
  const float* W1  = (const float*)d_in[2];
  const float* b1  = (const float*)d_in[3];
  const float* as1 = (const float*)d_in[4];
  const float* an1 = (const float*)d_in[5];
  const float* W2  = (const float*)d_in[6];
  const float* b2  = (const float*)d_in[7];
  const float* as2 = (const float*)d_in[8];
  const float* an2 = (const float*)d_in[9];
  float* out = (float*)d_out;

  char* ws = (char*)d_ws;
  size_t off = 0;
  auto alloc = [&](size_t bytes) -> void* {
    off = (off + 255) & ~(size_t)255;
    void* p = ws + off;
    off += bytes;
    return p;
  };

  _Float16* XH      = (_Float16*)alloc((size_t)N * NFEAT * 2);
  _Float16* W1T     = (_Float16*)alloc((size_t)H1 * NHID * NFEAT * 2);
  _Float16* W2T     = (_Float16*)alloc((size_t)NCLASS * H1 * NHID * 2);
  float*    feats1  = (float*)alloc((size_t)H1 * N * NHID * 4);
  _Float16* featsT  = (_Float16*)alloc((size_t)H1 * NHID * N * 2);
  float*    s1      = (float*)alloc((size_t)H1 * N * 4);
  float*    n1      = (float*)alloc((size_t)H1 * N * 4);
  float*    m1      = (float*)alloc((size_t)H1 * N * 4);
  float*    Z1      = (float*)alloc((size_t)H1 * N * 4);
  _Float16* H       = (_Float16*)alloc((size_t)N * H1 * NHID * 2);
  float*    feats2  = (float*)alloc((size_t)N * NCLASS * 4);
  _Float16* feats2T = (_Float16*)alloc((size_t)NCLASS * N * 2);
  float*    s2      = (float*)alloc((size_t)N * 4);
  float*    n2      = (float*)alloc((size_t)N * 4);
  float*    m2      = (float*)alloc((size_t)N * 4);
  float*    Z2      = (float*)alloc((size_t)N * 4);
  (void)ws_size; (void)n_in; (void)in_sizes; (void)out_size;

  // 0) casts + transposes
  {
    int total = N * NFEAT;  // largest of the three cast jobs
    cast_kernel<<<(total + 255) / 256, 256, 0, stream>>>(x, W1, W2, XH, W1T,
                                                         W2T);
  }
  // 1) per-head feature GEMM
  gemm1_kernel<<<dim3(N / 16, H1), 32, 0, stream>>>(XH, W1T, feats1, featsT);
  // 2) attention dot products
  sn_kernel<<<(H1 * N + 255) / 256, 256, 0, stream>>>(feats1, as1, an1, s1, n1,
                                                      NHID, N, H1 * N);
  // 3) softmax row stats (8 heads)
  rowstats_kernel<<<N, 256, 0, stream>>>(adj, s1, n1, m1, Z1, H1);
  // 4) fused attention GEMM + relu -> H (f16)
  attn_gemm1_kernel<<<dim3(N / 16, H1), 32, 0, stream>>>(adj, featsT, s1, n1,
                                                         m1, Z1, b1, H);
  // 5) layer-2 feature GEMM
  gemm2_kernel<<<N / 16, 32, 0, stream>>>(H, W2T, feats2, feats2T);
  // 6) layer-2 attention dots
  sn_kernel<<<(N + 255) / 256, 256, 0, stream>>>(feats2, as2, an2, s2, n2,
                                                 NCLASS, N, N);
  // 7) layer-2 row stats
  rowstats_kernel<<<N, 256, 0, stream>>>(adj, s2, n2, m2, Z2, 1);
  // 8) fused attention GEMM + relu + log-softmax
  attn_gemm2_kernel<<<N / 16, 32, 0, stream>>>(adj, feats2T, s2, n2, m2, Z2,
                                               b2, out);
}
